// BeamformingModule_67585605369982
// MI455X (gfx1250) — matrix operand order repaired and checked by reading, not compiled, not run on hardware
//
#include <hip/hip_runtime.h>

// ---------------------------------------------------------------------------
// Types for CDNA5 (gfx1250) WMMA
// ---------------------------------------------------------------------------
typedef __bf16 bf16_t;
typedef __attribute__((ext_vector_type(8)))  bf16_t v8bf;
typedef __attribute__((ext_vector_type(16))) bf16_t v16bf;
typedef __attribute__((ext_vector_type(8)))  float  v8f;

// ---------------------------------------------------------------------------
// Fused complex GEMM:  Yr = alpha*(Ar*Br^T - Ai*Bi^T) + br
//                      Yi = alpha*(Ar*Bi^T + Ai*Br^T) + bi
// A: (M,K) bf16 row-major (lda), B: (N,K) bf16 row-major (ldb) i.e. x @ W^T.
// Workgroup = 8 waves (wave32), each wave owns one 16x16 C tile of a 64x32
// workgroup tile. 4x v_wmma_f32_16x16x32_bf16 per K-step. No LDS: weights are
// tiny and L2(192MB)-resident; fragments load directly from global.
//
// Out-of-range-N lanes load from a clamped (row 0) address instead of being
// zero-masked: in the WMMA B layout column n lives only in lanes with
// lane%16==n, and D[:,n] lands in those same lanes, so garbage can only reach
// output columns the epilogue never stores. This keeps EXEC all-ones and the
// inner loop free of exec-mask traffic / fragment re-zeroing.
//
// Batched via blockIdx.z with linear A/B strides; C offset =
// (z/zdiv)*sCzHi + (z%zdiv)*sCzLo (for head-split epilogues).
// M must be a multiple of 64; K, lda, ldb multiples of 32 (buffers padded).
// ---------------------------------------------------------------------------
__global__ __launch_bounds__(256) void k_cgemm(
    const bf16_t* __restrict__ Ar, const bf16_t* __restrict__ Ai, long sAz,
    const bf16_t* __restrict__ Br, const bf16_t* __restrict__ Bi, long sBz,
    float* __restrict__ Cr, float* __restrict__ Ci,
    int zdiv, long sCzHi, long sCzLo,
    int M, int N, int K, int lda, int ldb, int ldc,
    float alpha, const float* __restrict__ biasR, const float* __restrict__ biasI)
{
  const int z = blockIdx.z;
  Ar += (long)z * sAz;  Ai += (long)z * sAz;
  Br += (long)z * sBz;  Bi += (long)z * sBz;
  const long co = (long)(z / zdiv) * sCzHi + (long)(z % zdiv) * sCzLo;
  Cr += co;  Ci += co;

  const int lane   = threadIdx.x & 31;
  const int w      = threadIdx.x >> 5;          // 0..7
  const int m0     = blockIdx.y * 64 + (w & 3) * 16;
  const int n0     = blockIdx.x * 32 + (w >> 2) * 16;
  const int l16    = lane & 15;
  const bool hiH   = (lane >= 16);

  // A fragment (16x32 bf16): lane<16 row M=l16 holds K {0..15};
  // lane>=16 holds K {16..31}.
  const long arow = (long)(m0 + l16) * lda + (hiH ? 16 : 0);
  // B fragment (32x16): lane%16 = N column; lane>=16 -> K base +16.
  const int  bn   = n0 + l16;
  const bool bok  = (bn < N);
  const long brow = (long)(bok ? bn : 0) * ldb + (hiH ? 16 : 0);

  v8f accP = {};  // Ar*Br
  v8f accM = {};  // Ai*Bi
  v8f accI = {};  // Ar*Bi + Ai*Br

  for (int k0 = 0; k0 < K; k0 += 32) {
    v8bf arLo = *(const v8bf*)(Ar + arow + k0);
    v8bf arHi = *(const v8bf*)(Ar + arow + k0 + 8);
    v8bf aiLo = *(const v8bf*)(Ai + arow + k0);
    v8bf aiHi = *(const v8bf*)(Ai + arow + k0 + 8);
    v16bf fAr = __builtin_shufflevector(arLo, arHi, 0,1,2,3,4,5,6,7,8,9,10,11,12,13,14,15);
    v16bf fAi = __builtin_shufflevector(aiLo, aiHi, 0,1,2,3,4,5,6,7,8,9,10,11,12,13,14,15);
    v16bf fBr = *(const v16bf*)(Br + brow + k0);
    v16bf fBi = *(const v16bf*)(Bi + brow + k0);
    accP = __builtin_amdgcn_wmma_f32_16x16x32_bf16(false, fAr, false, fBr, (short)0, accP, false, false);
    accM = __builtin_amdgcn_wmma_f32_16x16x32_bf16(false, fAi, false, fBi, (short)0, accM, false, false);
    accI = __builtin_amdgcn_wmma_f32_16x16x32_bf16(false, fAr, false, fBi, (short)0, accI, false, false);
    accI = __builtin_amdgcn_wmma_f32_16x16x32_bf16(false, fAi, false, fBr, (short)0, accI, false, false);
  }

  if (bok) {
    const float bR = biasR ? biasR[bn] : 0.f;
    const float bI = biasI ? biasI[bn] : 0.f;
    const int mrow = m0 + (hiH ? 8 : 0);
#pragma unroll
    for (int r = 0; r < 8; ++r) {
      const int mm = mrow + r;
      Cr[(long)mm * ldc + bn] = alpha * (accP[r] - accM[r]) + bR;
      Ci[(long)mm * ldc + bn] = alpha * accI[r] + bI;
    }
  }
}

// ---------------------------------------------------------------------------
// f32 -> bf16 staging with optional K zero-padding (ldd >= cols)
// ---------------------------------------------------------------------------
__global__ void k_cvt(const float* __restrict__ src, bf16_t* __restrict__ dst,
                      long rows, int cols, int lds, int ldd)
{
  long idx = (long)blockIdx.x * blockDim.x + threadIdx.x;
  long total = rows * (long)ldd;
  if (idx >= total) return;
  long r = idx / ldd;
  int  c = (int)(idx - r * ldd);
  float v = (c < cols) ? src[r * (long)lds + c] : 0.f;
  dst[idx] = (bf16_t)v;
}

// ---------------------------------------------------------------------------
// Block reductions (256 threads)
// ---------------------------------------------------------------------------
__device__ __forceinline__ float bsum(float v, float* sm) {
  int t = threadIdx.x;
  sm[t] = v; __syncthreads();
  for (int o = 128; o > 0; o >>= 1) { if (t < o) sm[t] += sm[t + o]; __syncthreads(); }
  float r = sm[0]; __syncthreads(); return r;
}
__device__ __forceinline__ float bmax(float v, float* sm) {
  int t = threadIdx.x;
  sm[t] = v; __syncthreads();
  for (int o = 128; o > 0; o >>= 1) { if (t < o) sm[t] = fmaxf(sm[t], sm[t + o]); __syncthreads(); }
  float r = sm[0]; __syncthreads(); return r;
}

// Complex layernorm over last dim (256). One block per row.
__global__ __launch_bounds__(256) void k_cln(const float* __restrict__ xr, const float* __restrict__ xi,
                                             float* __restrict__ outr, float* __restrict__ outi)
{
  __shared__ float sm[256];
  const long row = (long)blockIdx.x * 256;
  const int  t   = threadIdx.x;
  float r = xr[row + t], i = xi[row + t];
  const float mr = bsum(r, sm) * (1.f / 256.f);
  const float mi = bsum(i, sm) * (1.f / 256.f);
  r -= mr; i -= mi;
  const float vrr = bsum(r * r, sm) * (1.f / 256.f) + 1e-5f;
  const float vii = bsum(i * i, sm) * (1.f / 256.f) + 1e-5f;
  const float vri = bsum(r * i, sm) * (1.f / 256.f);
  const float s   = sqrtf(vrr * vii - vri * vri);
  const float tq  = sqrtf(vrr + vii + 2.f * s);
  const float inv = 1.f / (s * tq);
  const float wrr = (vii + s) * inv, wii = (vrr + s) * inv, wri = -vri * inv;
  outr[row + t] = wrr * r + wri * i;
  outi[row + t] = wri * r + wii * i;
}

// Row softmax (in place). One block per row.
__global__ __launch_bounds__(256) void k_softmax(float* __restrict__ x, int cols)
{
  __shared__ float sm[256];
  float* row = x + (long)blockIdx.x * cols;
  const int t = threadIdx.x;
  float mx = -3.4e38f;
  for (int c = t; c < cols; c += 256) mx = fmaxf(mx, row[c]);
  mx = bmax(mx, sm);
  float sum = 0.f;
  for (int c = t; c < cols; c += 256) { float e = __expf(row[c] - mx); row[c] = e; sum += e; }
  sum = bsum(sum, sm);
  const float inv = 1.f / sum;
  for (int c = t; c < cols; c += 256) row[c] *= inv;
}

// Elementwise helpers
__global__ void k_add(float* __restrict__ a, const float* __restrict__ b, long n) {
  long i = (long)blockIdx.x * blockDim.x + threadIdx.x;
  if (i < n) a[i] += b[i];
}
__global__ void k_leaky(float* __restrict__ x, long n) {
  long i = (long)blockIdx.x * blockDim.x + threadIdx.x;
  if (i < n) { float v = x[i]; x[i] = v > 0.f ? v : 0.01f * v; }
}

// Axis swap: (B, X, Y, D) -> (B, Y, X, D)
__global__ void k_tr(const float* __restrict__ in, float* __restrict__ out,
                     int Bn, int X, int Y, int Dm)
{
  long idx = (long)blockIdx.x * blockDim.x + threadIdx.x;
  long total = (long)Bn * X * Y * Dm;
  if (idx >= total) return;
  int d = (int)(idx % Dm); long r = idx / Dm;
  int y = (int)(r % Y); r /= Y;
  int x = (int)(r % X);
  int b = (int)(r / X);
  out[(((long)b * Y + y) * X + x) * Dm + d] = in[idx];
}

// Head split + bf16 stage, per batch b:  src (C,T,256) -> dst (C,H,T,64)
__global__ void k_pack(const float* __restrict__ sr, const float* __restrict__ si,
                       bf16_t* __restrict__ dr, bf16_t* __restrict__ di,
                       int Cn, int T, int H, int DK, int Dm)
{
  long idx = (long)blockIdx.x * blockDim.x + threadIdx.x;
  long total = (long)Cn * H * T * DK;
  if (idx >= total) return;
  int d = (int)(idx % DK); long r = idx / DK;
  int t = (int)(r % T); r /= T;
  int h = (int)(r % H);
  int c = (int)(r / H);
  long s = ((long)c * T + t) * Dm + h * DK + d;
  dr[idx] = (bf16_t)sr[s];
  di[idx] = (bf16_t)si[s];
}

// Head split transposed (for V):  src (C,T,256) -> dst (C,H,64,T)
__global__ void k_packT(const float* __restrict__ sr, const float* __restrict__ si,
                        bf16_t* __restrict__ dr, bf16_t* __restrict__ di,
                        int Cn, int T, int H, int DK, int Dm)
{
  long idx = (long)blockIdx.x * blockDim.x + threadIdx.x;
  long total = (long)Cn * H * DK * T;
  if (idx >= total) return;
  int t = (int)(idx % T); long r = idx / T;
  int d = (int)(r % DK); r /= DK;
  int h = (int)(r % H);
  int c = (int)(r / H);
  long s = ((long)c * T + t) * Dm + h * DK + d;
  dr[idx] = (bf16_t)sr[s];
  di[idx] = (bf16_t)si[s];
}

// Tiny masked complex attention over C=6 (layout (B,T,C,D)). One thread per
// (b,t,h,s). WMMA would be >60% padding at C=6, so VALU is the right tool.
__global__ void k_chattn(const float* __restrict__ qr, const float* __restrict__ qi,
                         const float* __restrict__ kr, const float* __restrict__ ki,
                         const float* __restrict__ vr, const float* __restrict__ vi,
                         const int* __restrict__ mask,
                         float* __restrict__ outr, float* __restrict__ outi,
                         int Bn, int T, int Cn, int H, int DK, int Dm)
{
  long idx = (long)blockIdx.x * blockDim.x + threadIdx.x;
  long total = (long)Bn * T * H * Cn;
  if (idx >= total) return;
  int s = (int)(idx % Cn); long r1 = idx / Cn;
  int h = (int)(r1 % H);  r1 /= H;
  int t = (int)(r1 % T);
  int b = (int)(r1 / T);
  const long rowbase = ((long)b * T + t) * Cn;
  const long qoff    = (rowbase + s) * Dm + h * DK;
  const float scale  = rsqrtf((float)DK);
  float sr[8], si[8];
  for (int tc = 0; tc < Cn; ++tc) {
    const long koff = (rowbase + tc) * Dm + h * DK;
    float ar = 0.f, ai = 0.f;
    for (int d = 0; d < DK; ++d) {
      float a = qr[qoff + d], bq = qi[qoff + d];
      float c = kr[koff + d], dd = ki[koff + d];
      ar += a * c - bq * dd;
      ai += a * dd + bq * c;
    }
    ar *= scale; ai *= scale;
    if (mask[((long)b * Cn + s) * Cn + tc] == 0) { ar = -1e9f; ai = -1e9f; }
    sr[tc] = ar; si[tc] = ai;
  }
  float mr = -3.4e38f, mi = -3.4e38f;
  for (int tc = 0; tc < Cn; ++tc) { mr = fmaxf(mr, sr[tc]); mi = fmaxf(mi, si[tc]); }
  float ssr = 0.f, ssi = 0.f;
  for (int tc = 0; tc < Cn; ++tc) {
    sr[tc] = __expf(sr[tc] - mr); ssr += sr[tc];
    si[tc] = __expf(si[tc] - mi); ssi += si[tc];
  }
  const float ir = 1.f / ssr, ii = 1.f / ssi;
  for (int tc = 0; tc < Cn; ++tc) { sr[tc] *= ir; si[tc] *= ii; }
  for (int d = 0; d < DK; ++d) {
    float orr = 0.f, oii = 0.f;
    for (int tc = 0; tc < Cn; ++tc) {
      const long voff = (rowbase + tc) * Dm + h * DK + d;
      float vrv = vr[voff], viv = vi[voff];
      orr += sr[tc] * vrv - si[tc] * viv;
      oii += sr[tc] * viv + si[tc] * vrv;
    }
    outr[qoff + d] = orr;
    outi[qoff + d] = oii;
  }
}

// ---------------------------------------------------------------------------
// Host orchestration
// ---------------------------------------------------------------------------
struct LinP {
  const float *wr, *wi, *br, *bi;
  int dout, din, dpad;
  bf16_t *bwr, *bwi;
};

extern "C" void kernel_launch(void* const* d_in, const int* in_sizes, int n_in,
                              void* d_out, int out_size, void* d_ws, size_t ws_size,
                              hipStream_t stream)
{
  (void)in_sizes; (void)n_in; (void)out_size; (void)ws_size;
  const int B = 8, C = 6, T = 512, F = 257, D = 256, H = 4, DK = 64, FFN = 2048;
  const int M = B * C * T;           // 24576
  const int FPAD = 288;              // 257 padded to mult of 32
  const int MC = 6144;               // FFN M-chunk (4 chunks)

  const float* x_r  = (const float*)d_in[0];
  const float* x_i  = (const float*)d_in[1];
  const int*   mask = (const int*)d_in[2];

  // ---- params (insertion-order flattening of the setup_inputs dict) ----
  int pi = 3;
  auto takeLin = [&](int dout, int din, int dpad) {
    LinP L;
    L.wr = (const float*)d_in[pi++]; L.wi = (const float*)d_in[pi++];
    L.br = (const float*)d_in[pi++]; L.bi = (const float*)d_in[pi++];
    L.dout = dout; L.din = din; L.dpad = dpad; L.bwr = nullptr; L.bwi = nullptr;
    return L;
  };
  LinP before = takeLin(D, F, FPAD);
  LinP after  = takeLin(F, D, D);
  struct Blk { LinP q, k, v, o, tq, tk, tv, to, f1, f2; } blks[2];
  for (int bl = 0; bl < 2; ++bl) {
    blks[bl].q  = takeLin(D, D, D);  blks[bl].k  = takeLin(D, D, D);
    blks[bl].v  = takeLin(D, D, D);  blks[bl].o  = takeLin(D, D, D);
    blks[bl].tq = takeLin(D, D, D);  blks[bl].tk = takeLin(D, D, D);
    blks[bl].tv = takeLin(D, D, D);  blks[bl].to = takeLin(D, D, D);
    blks[bl].f1 = takeLin(FFN, D, D);
    blks[bl].f2 = takeLin(D, FFN, FFN);
  }

  // ---- workspace carve-up ----
  char* ws = (char*)d_ws;
  size_t off = 0;
  auto alloc = [&](size_t bytes) -> char* {
    off = (off + 255) & ~(size_t)255;
    char* p = ws + off; off += bytes; return p;
  };
  const size_t actB = (size_t)M * D * sizeof(float);
  float* XR = (float*)alloc(actB); float* XI = (float*)alloc(actB);
  float* NR = (float*)alloc(actB); float* NI = (float*)alloc(actB);
  float* QR = (float*)alloc(actB); float* QI = (float*)alloc(actB);
  float* KR = (float*)alloc(actB); float* KI = (float*)alloc(actB);
  float* VR = (float*)alloc(actB); float* VI = (float*)alloc(actB);
  bf16_t* SAR = (bf16_t*)alloc((size_t)M * D * 2);
  bf16_t* SAI = (bf16_t*)alloc((size_t)M * D * 2);
  bf16_t* HB  = (bf16_t*)alloc((size_t)2 * MC * FFN * 2);   // hidden bf16 (r,i)
  const size_t pkN = (size_t)C * H * T * DK;
  bf16_t* QBR = (bf16_t*)alloc(pkN * 2); bf16_t* QBI = (bf16_t*)alloc(pkN * 2);
  bf16_t* KBR = (bf16_t*)alloc(pkN * 2); bf16_t* KBI = (bf16_t*)alloc(pkN * 2);
  bf16_t* VBR = (bf16_t*)alloc(pkN * 2); bf16_t* VBI = (bf16_t*)alloc(pkN * 2);
  char* big = alloc((size_t)2 * MC * FFN * sizeof(float));  // 100.7MB union scratch

  // ---- launch helpers ----
  auto cvt = [&](const float* src, bf16_t* dst, long rows, int cols, int lds, int ldd) {
    long n = rows * (long)ldd;
    k_cvt<<<dim3((unsigned)((n + 255) / 256)), 256, 0, stream>>>(src, dst, rows, cols, lds, ldd);
  };
  auto cgemm = [&](const bf16_t* Ar, const bf16_t* Ai, long sA, int lda,
                   const bf16_t* Br, const bf16_t* Bi, long sB, int ldb,
                   float* Cr, float* Ci, int zdiv, long hi, long lo, int ldc,
                   int m, int n, int k, int batch, float alpha,
                   const float* bR, const float* bI) {
    dim3 g((unsigned)((n + 31) / 32), (unsigned)((m + 63) / 64), (unsigned)batch);
    k_cgemm<<<g, 256, 0, stream>>>(Ar, Ai, sA, Br, Bi, sB, Cr, Ci, zdiv, hi, lo,
                                   m, n, k, lda, ldb, ldc, alpha, bR, bI);
  };
  auto stageW = [&](LinP& L) {
    L.bwr = (bf16_t*)alloc((size_t)L.dout * L.dpad * 2);
    L.bwi = (bf16_t*)alloc((size_t)L.dout * L.dpad * 2);
    cvt(L.wr, L.bwr, L.dout, L.din, L.din, L.dpad);
    cvt(L.wi, L.bwi, L.dout, L.din, L.din, L.dpad);
  };
  auto clin = [&](const bf16_t* ar, const bf16_t* ai, int lda, LinP& L,
                  float* cr, float* ci, int m) {
    cgemm(ar, ai, 0, lda, L.bwr, L.bwi, 0, L.dpad, cr, ci,
          1, 0, 0, L.dout, m, L.dout, L.dpad, 1, 1.f, L.br, L.bi);
  };
  auto add = [&](float* a, const float* b) {
    long n = (long)M * D;
    k_add<<<dim3((unsigned)((n + 255) / 256)), 256, 0, stream>>>(a, b, n);
  };

  // stage all weights to bf16 (L2-resident, ~13MB total)
  stageW(before); stageW(after);
  for (int bl = 0; bl < 2; ++bl) {
    stageW(blks[bl].q);  stageW(blks[bl].k);  stageW(blks[bl].v);  stageW(blks[bl].o);
    stageW(blks[bl].tq); stageW(blks[bl].tk); stageW(blks[bl].tv); stageW(blks[bl].to);
    stageW(blks[bl].f1); stageW(blks[bl].f2);
  }

  // ---- before-linear: (M,257)->(M,256) ----
  {
    bf16_t* XSR = (bf16_t*)big;
    bf16_t* XSI = XSR + (size_t)M * FPAD;
    cvt(x_r, XSR, M, F, F, FPAD);
    cvt(x_i, XSI, M, F, F, FPAD);
    clin(XSR, XSI, FPAD, before, XR, XI, M);
  }

  float *aXR = XR, *aXI = XI, *tNR = NR, *tNI = NI;
  auto swapf = [](float*& a, float*& b) { float* t = a; a = b; b = t; };

  for (int bl = 0; bl < 2; ++bl) {
    Blk& bk = blks[bl];

    // ===== time attention over T (params 'ch_att'), layout (B,C,T,D) =====
    k_cln<<<M, 256, 0, stream>>>(aXR, aXI, tNR, tNI);
    cvt(tNR, SAR, M, D, D, D);  cvt(tNI, SAI, M, D, D, D);
    clin(SAR, SAI, D, bk.q, QR, QI, M);
    clin(SAR, SAI, D, bk.k, KR, KI, M);
    clin(SAR, SAI, D, bk.v, VR, VI, M);

    float*  scR = (float*)big;
    float*  scI = scR + (size_t)C * H * T * T;
    bf16_t* pbR = (bf16_t*)(scI + (size_t)C * H * T * T);
    bf16_t* pbI = pbR + (size_t)C * H * T * T;
    const unsigned pkG = (unsigned)((pkN + 255) / 256);
    for (int b = 0; b < B; ++b) {
      const long base = (long)b * C * T * D;
      k_pack <<<pkG, 256, 0, stream>>>(QR + base, QI + base, QBR, QBI, C, T, H, DK, D);
      k_pack <<<pkG, 256, 0, stream>>>(KR + base, KI + base, KBR, KBI, C, T, H, DK, D);
      k_packT<<<pkG, 256, 0, stream>>>(VR + base, VI + base, VBR, VBI, C, T, H, DK, D);
      // scores: (T,T) per (c,h), alpha = 1/sqrt(64)
      cgemm(QBR, QBI, (long)T * DK, DK, KBR, KBI, (long)T * DK, DK,
            scR, scI, 1, (long)T * T, 0, T, T, T, DK, C * H, 0.125f, nullptr, nullptr);
      k_softmax<<<2 * C * H * T, 256, 0, stream>>>(scR, T);
      cvt(scR, pbR, (long)2 * C * H * T, T, T, T);
      // out = probs @ V  -> write merged heads into tN
      cgemm(pbR, pbI, (long)T * T, T, VBR, VBI, (long)DK * T, T,
            tNR + base, tNI + base, H, (long)T * D, DK, D,
            T, DK, T, C * H, 1.f, nullptr, nullptr);
    }
    cvt(tNR, SAR, M, D, D, D);  cvt(tNI, SAI, M, D, D, D);
    clin(SAR, SAI, D, bk.o, QR, QI, M);
    add(aXR, QR);  add(aXI, QI);

    // ===== swap to (B,T,C,D) =====
    {
      long n = (long)M * D;
      unsigned g = (unsigned)((n + 255) / 256);
      k_tr<<<g, 256, 0, stream>>>(aXR, tNR, B, C, T, D);
      k_tr<<<g, 256, 0, stream>>>(aXI, tNI, B, C, T, D);
      swapf(aXR, tNR); swapf(aXI, tNI);
    }

    // ===== channel attention over C with mask (params 'tm_att') =====
    k_cln<<<M, 256, 0, stream>>>(aXR, aXI, tNR, tNI);
    cvt(tNR, SAR, M, D, D, D);  cvt(tNI, SAI, M, D, D, D);
    clin(SAR, SAI, D, bk.tq, QR, QI, M);
    clin(SAR, SAI, D, bk.tk, KR, KI, M);
    clin(SAR, SAI, D, bk.tv, VR, VI, M);
    {
      long n = (long)B * T * H * C;
      k_chattn<<<dim3((unsigned)((n + 255) / 256)), 256, 0, stream>>>(
          QR, QI, KR, KI, VR, VI, mask, tNR, tNI, B, T, C, H, DK, D);
    }
    cvt(tNR, SAR, M, D, D, D);  cvt(tNI, SAI, M, D, D, D);
    clin(SAR, SAI, D, bk.to, QR, QI, M);
    add(aXR, QR);  add(aXI, QI);

    // ===== FFN (chunked over M to bound scratch) =====
    k_cln<<<M, 256, 0, stream>>>(aXR, aXI, tNR, tNI);
    cvt(tNR, SAR, M, D, D, D);  cvt(tNI, SAI, M, D, D, D);
    {
      float* HIDR = (float*)big;
      float* HIDI = HIDR + (size_t)MC * FFN;
      bf16_t* HBR = HB;
      bf16_t* HBI = HB + (size_t)MC * FFN;
      for (int ch = 0; ch < M / MC; ++ch) {
        const long r0 = (long)ch * MC;
        cgemm(SAR + r0 * D, SAI + r0 * D, 0, D, bk.f1.bwr, bk.f1.bwi, 0, D,
              HIDR, HIDI, 1, 0, 0, FFN, MC, FFN, D, 1, 1.f, bk.f1.br, bk.f1.bi);
        long hn = (long)2 * MC * FFN;
        k_leaky<<<dim3((unsigned)((hn + 255) / 256)), 256, 0, stream>>>(HIDR, hn);
        cvt(HIDR, HBR, (long)2 * MC, FFN, FFN, FFN);
        cgemm(HBR, HBI, 0, FFN, bk.f2.bwr, bk.f2.bwi, 0, FFN,
              QR + r0 * D, QI + r0 * D, 1, 0, 0, D, MC, D, FFN, 1, 1.f,
              bk.f2.br, bk.f2.bi);
      }
    }
    add(aXR, QR);  add(aXI, QI);

    // ===== swap back to (B,C,T,D) =====
    {
      long n = (long)M * D;
      unsigned g = (unsigned)((n + 255) / 256);
      k_tr<<<g, 256, 0, stream>>>(aXR, tNR, B, T, C, D);
      k_tr<<<g, 256, 0, stream>>>(aXI, tNI, B, T, C, D);
      swapf(aXR, tNR); swapf(aXI, tNI);
    }
  }

  // ---- after-linear: (M,256)->(M,257), straight into d_out (xr then xi) ----
  cvt(aXR, SAR, M, D, D, D);  cvt(aXI, SAI, M, D, D, D);
  {
    float* outR = (float*)d_out;
    float* outI = outR + (size_t)M * F;
    clin(SAR, SAI, D, after, outR, outI, M);
  }
}
